// QuantLinearWrapper_16269336117399
// MI455X (gfx1250) — 8-bit matrix operand order repaired and checked by reading, compiled, not run on hardware
//
#include <hip/hip_runtime.h>

typedef __attribute__((ext_vector_type(8))) int v8i;
typedef __attribute__((ext_vector_type(4))) int v4i;
typedef __attribute__((ext_vector_type(2))) int v2i;

#define MTILE 128
#define NTILE 128
#define KSTEP 64
#define LDS_STRIDE 80

#if __has_builtin(__builtin_amdgcn_global_load_async_to_lds_b128) && \
    __has_builtin(__builtin_amdgcn_s_wait_asynccnt)
#define USE_ASYNC_LDS 1
#else
#define USE_ASYNC_LDS 0
#endif

#if USE_ASYNC_LDS
typedef __attribute__((address_space(1))) v4i as1_v4i;
typedef __attribute__((address_space(3))) v4i as3_v4i;

__device__ __forceinline__ void cp16_async(void* lds, const void* g) {
    __builtin_amdgcn_global_load_async_to_lds_b128(
        (as1_v4i*)g, (as3_v4i*)lds,  0,  0);
}
#endif


__global__ void init_ws_kernel(unsigned* am) {
    am[0] = 0u;
    am[1] = 0u;
}

__global__ void absmax_kernel(const float* __restrict__ in, size_t n,
                              unsigned* __restrict__ out) {
    unsigned m = 0u;
    for (size_t i = (size_t)blockIdx.x * blockDim.x + threadIdx.x; i < n;
         i += (size_t)gridDim.x * blockDim.x) {
        m = max(m, __float_as_uint(fabsf(in[i])));
    }
    for (int off = 16; off > 0; off >>= 1)
        m = max(m, (unsigned)__shfl_xor((int)m, off, 32));
    __shared__ unsigned sm[8];
    int lane = threadIdx.x & 31;
    int wid  = threadIdx.x >> 5;
    if (lane == 0) sm[wid] = m;
    __syncthreads();
    if (threadIdx.x == 0) {
        unsigned r = sm[0];
        int nw = blockDim.x >> 5;
        for (int w = 1; w < nw; ++w) r = max(r, sm[w]);
        atomicMax(out, r);
    }
}

__device__ __forceinline__ float sym_scale_from_bits(unsigned bits) {
    return fmaxf(__uint_as_float(bits), 1e-8f) * (1.0f / 127.0f);
}

__global__ void quant8_kernel(const float4* __restrict__ in,
                              char* __restrict__ out, size_t n4,
                              const unsigned* __restrict__ absmax) {
    float scale = sym_scale_from_bits(*absmax);
    size_t i = (size_t)blockIdx.x * blockDim.x + threadIdx.x;
    if (i >= n4) return;
    float4 v = in[i];
    float q0 = fminf(fmaxf(rintf(v.x / scale), -128.0f), 127.0f);
    float q1 = fminf(fmaxf(rintf(v.y / scale), -128.0f), 127.0f);
    float q2 = fminf(fmaxf(rintf(v.z / scale), -128.0f), 127.0f);
    float q3 = fminf(fmaxf(rintf(v.w / scale), -128.0f), 127.0f);
    int packed = ((int)q0 & 0xff) | (((int)q1 & 0xff) << 8) |
                 (((int)q2 & 0xff) << 16) | (((int)q3 & 0xff) << 24);
    ((int*)out)[i] = packed;
}

__global__ void quant_bias_kernel(const float* __restrict__ b,
                                  int* __restrict__ b_int, int n,
                                  const unsigned* __restrict__ amx,
                                  const unsigned* __restrict__ amw) {
    float bscale = sym_scale_from_bits(*amx) * sym_scale_from_bits(*amw);
    int i = blockIdx.x * blockDim.x + threadIdx.x;
    if (i >= n) return;
    float q = rintf(b[i] / bscale);
    q = fminf(fmaxf(q, -2147483648.0f), 2147483520.0f);
    b_int[i] = (int)q;
}

__global__ __launch_bounds__(256)
void gemm_i8_wmma_kernel(const char* __restrict__ A,
                         const char* __restrict__ Bm,
                         const int*  __restrict__ bias_int,
                         float* __restrict__ out,
                         int M, int N, int K,
                         const unsigned* __restrict__ amx,
                         const unsigned* __restrict__ amw) {
    float bscale = sym_scale_from_bits(*amx) * sym_scale_from_bits(*amw);

    __shared__ __align__(16) char As[2][MTILE * LDS_STRIDE];
    __shared__ __align__(16) char Bs[2][NTILE * LDS_STRIDE];

    const int tid    = threadIdx.x;
    const int wave   = tid >> 5;
    const int lane   = tid & 31;
    const int wm     = wave >> 1;
    const int wn     = wave & 1;
    const int laneM  = lane & 15;
    const int laneHi = lane >> 4;

    const int blockM = blockIdx.y * MTILE;
    const int blockN = blockIdx.x * NTILE;

    const int idx0 = tid;
    const int idx1 = tid + 256;
    const int r0 = idx0 >> 2, c0 = (idx0 & 3) << 4;
    const int r1 = idx1 >> 2, c1 = (idx1 & 3) << 4;
    const char* gA0 = A  + (size_t)(blockM + r0) * K + c0;
    const char* gA1 = A  + (size_t)(blockM + r1) * K + c1;
    const char* gB0 = Bm + (size_t)(blockN + r0) * K + c0;
    const char* gB1 = Bm + (size_t)(blockN + r1) * K + c1;
    const int l0 = r0 * LDS_STRIDE + c0;
    const int l1 = r1 * LDS_STRIDE + c1;

    v8i acc[2][4];
    const v8i vzero = {0, 0, 0, 0, 0, 0, 0, 0};
    for (int mi = 0; mi < 2; ++mi)
        for (int ni = 0; ni < 4; ++ni)
            acc[mi][ni] = vzero;

    const int nk = K / KSTEP;

#if USE_ASYNC_LDS
    cp16_async(As[0] + l0, gA0);
    cp16_async(As[0] + l1, gA1);
    cp16_async(Bs[0] + l0, gB0);
    cp16_async(Bs[0] + l1, gB1);
    __builtin_amdgcn_s_wait_asynccnt(0);
#else
    *(v4i*)(As[0] + l0) = *(const v4i*)(gA0);
    *(v4i*)(As[0] + l1) = *(const v4i*)(gA1);
    *(v4i*)(Bs[0] + l0) = *(const v4i*)(gB0);
    *(v4i*)(Bs[0] + l1) = *(const v4i*)(gB1);
#endif
    __syncthreads();

    for (int ik = 0; ik < nk; ++ik) {
        const int cur = ik & 1;
        const int nxt = cur ^ 1;
        const int ktn = (ik + 1) * KSTEP;
        const bool have_next = (ik + 1 < nk);

#if USE_ASYNC_LDS
        if (have_next) {
            cp16_async(As[nxt] + l0, gA0 + ktn);
            cp16_async(As[nxt] + l1, gA1 + ktn);
            cp16_async(Bs[nxt] + l0, gB0 + ktn);
            cp16_async(Bs[nxt] + l1, gB1 + ktn);
        }
#else
        v4i tA0, tA1, tB0, tB1;
        if (have_next) {
            tA0 = *(const v4i*)(gA0 + ktn);
            tA1 = *(const v4i*)(gA1 + ktn);
            tB0 = *(const v4i*)(gB0 + ktn);
            tB1 = *(const v4i*)(gB1 + ktn);
        }
#endif

        v8i afrag[2];
        #pragma unroll
        for (int mi = 0; mi < 2; ++mi) {
            const char* ap = As[cur] + (wm * 32 + mi * 16 + laneM) * LDS_STRIDE + laneHi * 16;
            v2i p0 = *(const v2i*)(ap);
            v2i p1 = *(const v2i*)(ap + 8);
            v2i p2 = *(const v2i*)(ap + 32);
            v2i p3 = *(const v2i*)(ap + 40);
            v8i a = {p0.x, p0.y, p1.x, p1.y, p2.x, p2.y, p3.x, p3.y};
            afrag[mi] = a;
        }
        v8i bfrag[4];
        #pragma unroll
        for (int ni = 0; ni < 4; ++ni) {
            const char* bp = Bs[cur] + (wn * 64 + ni * 16 + laneM) * LDS_STRIDE + laneHi * 16;
            v4i q0 = *(const v4i*)(bp);
            v4i q1 = *(const v4i*)(bp + 32);
            v8i bb = {q0.x, q0.y, q0.z, q0.w, q1.x, q1.y, q1.z, q1.w};
            bfrag[ni] = bb;
        }

        #pragma unroll
        for (int mi = 0; mi < 2; ++mi)
            #pragma unroll
            for (int ni = 0; ni < 4; ++ni)
                acc[mi][ni] = __builtin_amdgcn_wmma_i32_16x16x64_iu8(
                     true, afrag[mi],
                     true, bfrag[ni],
                    acc[mi][ni],  false,  false);

#if USE_ASYNC_LDS
        __builtin_amdgcn_s_wait_asynccnt(0);
#else
        if (have_next) {
            *(v4i*)(As[nxt] + l0) = tA0;
            *(v4i*)(As[nxt] + l1) = tA1;
            *(v4i*)(Bs[nxt] + l0) = tB0;
            *(v4i*)(Bs[nxt] + l1) = tB1;
        }
#endif
        __syncthreads();
    }

    #pragma unroll
    for (int mi = 0; mi < 2; ++mi) {
        #pragma unroll
        for (int ni = 0; ni < 4; ++ni) {
            int col = blockN + wn * 64 + ni * 16 + laneM;
            int bi  = bias_int[col];
            #pragma unroll
            for (int r = 0; r < 8; ++r) {
                int row = blockM + wm * 32 + mi * 16 + laneHi * 8 + r;
                float v = (float)(acc[mi][ni][r] + bi) * bscale;
                __builtin_nontemporal_store(v, &out[(size_t)row * N + col]);
            }
        }
    }
}

extern "C" void kernel_launch(void* const* d_in, const int* in_sizes, int n_in,
                              void* d_out, int out_size, void* d_ws, size_t ws_size,
                              hipStream_t stream) {
    const float* x = (const float*)d_in[0];
    const float* W = (const float*)d_in[1];
    const float* b = (const float*)d_in[2];
    float* out = (float*)d_out;

    const int N = in_sizes[2];
    const int K = in_sizes[1] / N;
    const int M = (int)(in_sizes[0] / K);

    char* wsb = (char*)d_ws;
    unsigned* amx  = (unsigned*)(wsb + 0);
    unsigned* amw  = (unsigned*)(wsb + 4);
    int* b_int     = (int*)(wsb + 256);
    char* x_q      = wsb + 32768;
    char* w_q      = wsb + 32768 + (size_t)M * K;

    init_ws_kernel<<<1, 1, 0, stream>>>(amx);

    const size_t nX = (size_t)M * K;
    const size_t nW = (size_t)N * K;
    absmax_kernel<<<2048, 256, 0, stream>>>(x, nX, amx);
    absmax_kernel<<<1024, 256, 0, stream>>>(W, nW, amw);

    const size_t nX4 = nX / 4, nW4 = nW / 4;
    quant8_kernel<<<(unsigned)((nX4 + 255) / 256), 256, 0, stream>>>(
        (const float4*)x, x_q, nX4, amx);
    quant8_kernel<<<(unsigned)((nW4 + 255) / 256), 256, 0, stream>>>(
        (const float4*)W, w_q, nW4, amw);
    quant_bias_kernel<<<(N + 255) / 256, 256, 0, stream>>>(b, b_int, N, amx, amw);

    dim3 grid(N / NTILE, M / MTILE);
    gemm_i8_wmma_kernel<<<grid, 256, 0, stream>>>(x_q, w_q, b_int, out,
                                                  M, N, K, amx, amw);
}
